// QGRUCell_38749194945208
// MI455X (gfx1250) — matrix operand order repaired and checked by reading, not compiled, not run on hardware
//
#include <hip/hip_runtime.h>
#include <cstdint>
#include <cstddef>

typedef __attribute__((ext_vector_type(8)))  float        v8f;
typedef __attribute__((ext_vector_type(16))) __bf16       v16bf;
typedef __attribute__((ext_vector_type(8)))  unsigned int v8u;

// ---------------------------------------------------------------------------
// bf16 packing helpers (setup kernel builds pre-swizzled WMMA B fragments)
// ---------------------------------------------------------------------------
static __device__ __forceinline__ unsigned int pkbf(float a, float b) {
    union { __bf16 h; unsigned short s; } ua, ub;
    ua.h = (__bf16)a; ub.h = (__bf16)b;
    return (unsigned int)ua.s | ((unsigned int)ub.s << 16);
}

// Pass-1 packed B matrix (2048 x 16): cols 0-3 = Wu_in, 4-7 = Wr_in,
// 8-11 = Wc_in (x-part only, zero for k>=1024), 12-15 = 0.
static __device__ __forceinline__ float bval1(const float* Wu, const float* Wr,
                                              const float* Wc, int k, int n) {
    if (n < 4)  return Wu[n * 2048 + k];
    if (n < 8)  return Wr[(n - 4) * 2048 + k];
    if (n < 12) return (k < 1024) ? Wc[(n - 8) * 2048 + k] : 0.0f;
    return 0.0f;
}

// Pre-swizzle B into exact 16-bit B-fragment layout (CDNA5 ISA):
// B 32x16 bf16: lane n(0-15): VGPR v<4 holds K = 2v,2v+1, v>=4: K = 16+2(v-4),+1; lane n+16: same +8.
__global__ void qgru_pack(const float* __restrict__ Wu_in,
                          const float* __restrict__ Wr_in,
                          const float* __restrict__ Wc_in,
                          unsigned int* __restrict__ bp1,
                          unsigned int* __restrict__ bp2) {
    int t    = blockIdx.x * blockDim.x + threadIdx.x;
    int lane = t & 31, blk = t >> 5;
    int col  = lane & 15;
    int khalf = (lane >> 4) * 8;
    if (blk < 64) {                       // pass-1 table: K = 0..2047, 64 blocks of 32
        unsigned int* dst = bp1 + (size_t)(blk * 32 + lane) * 8;
        #pragma unroll
        for (int d = 0; d < 8; ++d) {
            int k = blk * 32 + khalf + 2 * (d & 3) + (d >> 2) * 16;
            dst[d] = pkbf(bval1(Wu_in, Wr_in, Wc_in, k,     col),
                          bval1(Wu_in, Wr_in, Wc_in, k + 1, col));
        }
    } else if (blk < 96) {                // pass-2 table: h-part of Wc_in, K = 0..1023
        int b2 = blk - 64;
        unsigned int* dst = bp2 + (size_t)(b2 * 32 + lane) * 8;
        #pragma unroll
        for (int d = 0; d < 8; ++d) {
            int k = b2 * 32 + khalf + 2 * (d & 3) + (d >> 2) * 16;
            float v0 = (col < 4) ? Wc_in[col * 2048 + 1024 + k]     : 0.0f;
            float v1 = (col < 4) ? Wc_in[col * 2048 + 1024 + k + 1] : 0.0f;
            dst[d] = pkbf(v0, v1);
        }
    }
}

// ---------------------------------------------------------------------------
// 4-qubit statevector simulator (one full sim per lane; 16 complex amps)
// state index = q0*8 + q1*4 + q2*2 + q3   (wire w -> bit 3-w)
// ---------------------------------------------------------------------------
struct QState { float re[16]; float im[16]; };

static __device__ __forceinline__ void apply1q(QState& s, int w,
        float u00r, float u00i, float u01r, float u01i,
        float u10r, float u10i, float u11r, float u11i) {
    const int mask = 1 << (3 - w);
    #pragma unroll
    for (int i = 0; i < 16; ++i) {
        if (i & mask) continue;
        const int j = i | mask;
        float ar = s.re[i], ai = s.im[i], br = s.re[j], bi = s.im[j];
        s.re[i] = u00r * ar - u00i * ai + u01r * br - u01i * bi;
        s.im[i] = u00r * ai + u00i * ar + u01r * bi + u01i * br;
        s.re[j] = u10r * ar - u10i * ai + u11r * br - u11i * bi;
        s.im[j] = u10r * ai + u10i * ar + u11r * bi + u11i * br;
    }
}

static __device__ __forceinline__ void apply_cnot(QState& s, int c, int t) {
    const int bc = 3 - c, bt = 3 - t;
    #pragma unroll
    for (int i = 0; i < 16; ++i) {
        if (((i >> bc) & 1) && !((i >> bt) & 1)) {
            const int j = i | (1 << bt);
            float tr = s.re[i]; s.re[i] = s.re[j]; s.re[j] = tr;
            float ti = s.im[i]; s.im[i] = s.im[j]; s.im[j] = ti;
        }
    }
}

static __device__ void qsim(const float a[4], const float* __restrict__ qw, float q[4]) {
    QState s;
    #pragma unroll
    for (int i = 0; i < 16; ++i) { s.re[i] = 0.0f; s.im[i] = 0.0f; }
    s.re[0] = 1.0f;
    // AngleEmbedding: RX(a[w]) on each wire
    #pragma unroll
    for (int w = 0; w < 4; ++w) {
        float sn, cs; __sincosf(0.5f * a[w], &sn, &cs);
        apply1q(s, w, cs, 0.0f, 0.0f, -sn, 0.0f, -sn, cs, 0.0f);
    }
    // StronglyEntanglingLayers: Rot + ring CNOTs, ranges = (l % 3) + 1
    #pragma unroll
    for (int l = 0; l < 2; ++l) {
        #pragma unroll
        for (int w = 0; w < 4; ++w) {
            float phi = qw[(l * 4 + w) * 3 + 0];
            float th  = qw[(l * 4 + w) * 3 + 1];
            float om  = qw[(l * 4 + w) * 3 + 2];
            float st, ct; __sincosf(0.5f * th, &st, &ct);
            float sp, cp; __sincosf(-0.5f * (phi + om), &sp, &cp); // e^{-i(phi+om)/2}
            float sm, cm; __sincosf( 0.5f * (phi - om), &sm, &cm); // e^{+i(phi-om)/2}
            apply1q(s, w, cp * ct,  sp * ct,
                          -cm * st, -sm * st,
                           cm * st, -sm * st,
                           cp * ct, -sp * ct);
        }
        const int r = (l % 3) + 1;
        #pragma unroll
        for (int w = 0; w < 4; ++w) apply_cnot(s, w, (w + r) & 3);
    }
    // PauliZ expectation values
    #pragma unroll
    for (int w = 0; w < 4; ++w) {
        const int bit = 3 - w;
        float acc = 0.0f;
        #pragma unroll
        for (int i = 0; i < 16; ++i) {
            float p = s.re[i] * s.re[i] + s.im[i] * s.im[i];
            acc += ((i >> bit) & 1) ? -p : p;
        }
        q[w] = acc;
    }
}

// ---------------------------------------------------------------------------
// Main fused kernel: one wave per 16-row batch tile.
// ---------------------------------------------------------------------------
static __device__ __forceinline__ float sigm(float x) { return 1.0f / (1.0f + __expf(-x)); }

static __device__ __forceinline__ void setA8(v16bf& f, int base, float4 a, float4 b) {
    f[base + 0] = (__bf16)a.x; f[base + 1] = (__bf16)a.y;
    f[base + 2] = (__bf16)a.z; f[base + 3] = (__bf16)a.w;
    f[base + 4] = (__bf16)b.x; f[base + 5] = (__bf16)b.y;
    f[base + 6] = (__bf16)b.z; f[base + 7] = (__bf16)b.w;
}

__global__ __launch_bounds__(256) void qgru_main(
        const float* __restrict__ xp,     const float* __restrict__ hp,
        const float* __restrict__ bu_in,  const float* __restrict__ qwu,
        const float* __restrict__ Wu_out, const float* __restrict__ bu_out,
        const float* __restrict__ br_in,  const float* __restrict__ qwr,
        const float* __restrict__ Wr_out, const float* __restrict__ br_out,
        const float* __restrict__ bc_in,  const float* __restrict__ qwc,
        const float* __restrict__ Wc_out, const float* __restrict__ bc_out,
        const unsigned int* __restrict__ bp1, const unsigned int* __restrict__ bp2,
        float* __restrict__ out, int ntiles) {
    __shared__ float angA[8][16][16];   // pass-1 C fragment exchange
    __shared__ float angB[8][16][16];   // pass-2 C fragment exchange
    __shared__ float qU[8][16][4];
    __shared__ float qR[8][16][4];
    __shared__ float qC[8][16][4];

    const int wave = threadIdx.x >> 5;
    const int lane = threadIdx.x & 31;
    const int row  = lane & 15;          // A: M-row / C: N-col held by this lane
    const int half = lane >> 4;
    int tile = blockIdx.x * 8 + wave;
    if (tile >= ntiles) tile = ntiles - 1;          // keep EXEC all-ones (benign dup)
    const size_t rbase = (size_t)(tile * 16 + row) * 1024;

    // ---- Pass 1: angles(u,r) + x-part of c.  comb = [x | h_prev], K = 2048 ----
    // 16-bit A fragment: lanes 0-15 hold K runs [k0, k0+8) & [k0+16, k0+24),
    // lanes 16-31 hold [k0+8, k0+16) & [k0+24, k0+32) of their row.
    v8f acc = {};
    #pragma unroll 4
    for (int t = 0; t < 64; ++t) {
        const int kk1 = t * 32 + half * 8;
        const int kk2 = kk1 + 16;
        const float* s1 = (kk1 < 1024) ? (xp + rbase + kk1) : (hp + rbase + kk1 - 1024);
        const float* s2 = (kk2 < 1024) ? (xp + rbase + kk2) : (hp + rbase + kk2 - 1024);
        __builtin_prefetch(s1 + 32, 0, 0);          // global_prefetch_b8, next k-block
        v16bf af;
        setA8(af, 0, *(const float4*)s1, *(const float4*)(s1 + 4));
        setA8(af, 8, *(const float4*)s2, *(const float4*)(s2 + 4));
        union { v8u u; v16bf b; } bf;
        bf.u = *(const v8u*)(bp1 + (size_t)(t * 32 + lane) * 8);
        acc = __builtin_amdgcn_wmma_f32_16x16x32_bf16(false, af, false, bf.b,
                                                      (short)0, acc, false, false);
    }
    #pragma unroll
    for (int v = 0; v < 8; ++v) angA[wave][half * 8 + v][row] = acc[v];
    __syncthreads();

    // ---- quantum sims: lanes 0-15 -> gate u, lanes 16-31 -> gate r ----
    {
        const float* bin = half ? br_in : bu_in;
        const float* qwp = half ? qwr   : qwu;
        float a[4], q[4];
        #pragma unroll
        for (int j = 0; j < 4; ++j) a[j] = angA[wave][row][half * 4 + j] + bin[j];
        qsim(a, qwp, q);
        float (*dst)[16][4] = half ? qR : qU;
        #pragma unroll
        for (int j = 0; j < 4; ++j) dst[wave][row][j] = q[j];
    }
    __syncthreads();

    // ---- Pass 2: h-part of c angles: sum_k sigmoid(q_r . Wr_out[k]) * h[k] * Wc_in[:,1024+k]
    const float qr0 = qR[wave][row][0], qr1 = qR[wave][row][1];
    const float qr2 = qR[wave][row][2], qr3 = qR[wave][row][3];
    v8f acc2 = {};
    #pragma unroll 2
    for (int t = 0; t < 32; ++t) {
        const int kk1 = t * 32 + half * 8;
        v16bf af;
        #pragma unroll
        for (int rb = 0; rb < 2; ++rb) {
            const int kb = kk1 + rb * 16;
            float4 h0 = *(const float4*)(hp + rbase + kb);
            float4 h1 = *(const float4*)(hp + rbase + kb + 4);
            float hv[8] = {h0.x, h0.y, h0.z, h0.w, h1.x, h1.y, h1.z, h1.w};
            #pragma unroll
            for (int e = 0; e < 8; ++e) {
                const int k = kb + e;
                float4 w4 = *(const float4*)(Wr_out + (size_t)k * 4);
                float rr = sigm(qr0 * w4.x + qr1 * w4.y + qr2 * w4.z + qr3 * w4.w
                                + br_out[k]);
                af[rb * 8 + e] = (__bf16)(rr * hv[e]);
            }
        }
        union { v8u u; v16bf b; } bf;
        bf.u = *(const v8u*)(bp2 + (size_t)(t * 32 + lane) * 8);
        acc2 = __builtin_amdgcn_wmma_f32_16x16x32_bf16(false, af, false, bf.b,
                                                       (short)0, acc2, false, false);
    }
    #pragma unroll
    for (int v = 0; v < 8; ++v) angB[wave][half * 8 + v][row] = acc2[v];
    __syncthreads();

    // ---- quantum sim for gate c (lanes 0-15) ----
    if (half == 0) {
        float a[4], q[4];
        #pragma unroll
        for (int j = 0; j < 4; ++j)
            a[j] = angA[wave][row][8 + j] + angB[wave][row][j] + bc_in[j];
        qsim(a, qwc, q);
        #pragma unroll
        for (int j = 0; j < 4; ++j) qC[wave][row][j] = q[j];
    }
    __syncthreads();

    // ---- Final: h = (1-z)*h_prev + z*tanh(c_out), fully coalesced ----
    const size_t tbase = (size_t)(tile * 16) * 1024;
    for (int idx = lane; idx < 16 * 1024; idx += 32) {
        const int rr = idx >> 10;        // uniform across the wave each iteration
        const int k  = idx & 1023;
        float4 wu = *(const float4*)(Wu_out + (size_t)k * 4);
        float4 wc = *(const float4*)(Wc_out + (size_t)k * 4);
        float z = sigm(qU[wave][rr][0] * wu.x + qU[wave][rr][1] * wu.y +
                       qU[wave][rr][2] * wu.z + qU[wave][rr][3] * wu.w + bu_out[k]);
        float ht = tanhf(qC[wave][rr][0] * wc.x + qC[wave][rr][1] * wc.y +
                         qC[wave][rr][2] * wc.z + qC[wave][rr][3] * wc.w + bc_out[k]);
        float hv = hp[tbase + (size_t)rr * 1024 + k];
        out[tbase + (size_t)rr * 1024 + k] = (1.0f - z) * hv + z * ht;
    }
}

// ---------------------------------------------------------------------------
extern "C" void kernel_launch(void* const* d_in, const int* in_sizes, int n_in,
                              void* d_out, int out_size, void* d_ws, size_t ws_size,
                              hipStream_t stream) {
    const float* x      = (const float*)d_in[0];
    const float* h      = (const float*)d_in[1];
    const float* Wu_in  = (const float*)d_in[2];
    const float* bu_in  = (const float*)d_in[3];
    const float* qwu    = (const float*)d_in[4];
    const float* Wu_out = (const float*)d_in[5];
    const float* bu_out = (const float*)d_in[6];
    const float* Wr_in  = (const float*)d_in[7];
    const float* br_in  = (const float*)d_in[8];
    const float* qwr    = (const float*)d_in[9];
    const float* Wr_out = (const float*)d_in[10];
    const float* br_out = (const float*)d_in[11];
    const float* Wc_in  = (const float*)d_in[12];
    const float* bc_in  = (const float*)d_in[13];
    const float* qwc    = (const float*)d_in[14];
    const float* Wc_out = (const float*)d_in[15];
    const float* bc_out = (const float*)d_in[16];

    const int B      = in_sizes[0] / 1024;  // 16384
    const int ntiles = B / 16;              // 1024 tiles of 16 rows

    unsigned int* bp1 = (unsigned int*)d_ws;          // 64 blk * 32 lane * 8 dw = 64 KB
    unsigned int* bp2 = bp1 + 64 * 32 * 8;            // 32 blk * 32 lane * 8 dw = 32 KB

    qgru_pack<<<12, 256, 0, stream>>>(Wu_in, Wr_in, Wc_in, bp1, bp2);

    const int blocks = (ntiles + 7) / 8;              // 8 waves (tiles) per block
    qgru_main<<<blocks, 256, 0, stream>>>(
        x, h,
        bu_in, qwu, Wu_out, bu_out,
        br_in, qwr, Wr_out, br_out,
        bc_in, qwc, Wc_out, bc_out,
        bp1, bp2, (float*)d_out, ntiles);
}
